// SelfAttention_3453153706579
// MI455X (gfx1250) — matrix operand order repaired and checked by reading, not compiled, not run on hardware
//
#include <hip/hip_runtime.h>
#include <hip/hip_bf16.h>

// ---------------------------------------------------------------------------
// Self-attention, B=2 N=2048 H=1024 NH=16 DH=64, scale = H^-0.5 = 1/32.
// Kernel 1: QKV projection (bf16 WMMA, fp32 accumulate), 128x128 block tile,
//           8 waves x (32x64), writes Q*scale,K,V bf16 in [b,h,n,d] layout.
// Kernel 2: flash attention with online softmax; K tiles staged into LDS via
//           the Tensor Data Mover (tensor_load_to_lds + s_wait_tensorcnt),
//           bf16 WMMA for QK^T and P*V, fp32 output.
// ---------------------------------------------------------------------------

typedef __attribute__((ext_vector_type(16))) __bf16 v16bf;
typedef __attribute__((ext_vector_type(8)))  float  v8f;
typedef __attribute__((ext_vector_type(4)))  unsigned u32x4;
typedef __attribute__((ext_vector_type(8)))  int      i32x8;
typedef __attribute__((ext_vector_type(4)))  int      i32x4;

union FragAB { unsigned u[8]; v16bf v; };

__device__ __forceinline__ unsigned short f2bf(float f) {
    unsigned u = __float_as_uint(f);
    u += 0x7FFFu + ((u >> 16) & 1u);      // round-to-nearest-even
    return (unsigned short)(u >> 16);
}
__device__ __forceinline__ unsigned pack2bf(float a, float b) {
    return (unsigned)f2bf(a) | ((unsigned)f2bf(b) << 16);
}

// A-matrix (16-bit, 16x32) per-lane dword offset within a 32-wide K chunk:
// lanes 0-15:  V0..3 -> K(0,1..6,7), V4..7 -> K(16,17..22,23)
// lanes 16-31: V0..3 -> K(8..15),    V4..7 -> K(24..31)
__device__ __forceinline__ int a_kdw(int v, int hi) {
    return ((v & 4) ? (v + 4) : v) + 4 * hi;   // dword index (element k / 2)
}
// B-matrix (16-bit, 32x16): same K order as A (lanes0-15 V0..3 K0..7, V4..7 K16..23; lanes16-31 K8..15, K24..31)
__device__ __forceinline__ int b_kdw(int v, int hi) {
    return ((v & 4) ? (v + 4) : v) + 4 * hi;   // dword index
}

// ---------------------------------------------------------------------------
// Kernel 1: C[4096,3072] = X[4096,1024] * W^T,  W is [3072,1024] row-major.
// Workgroup tile 128(M) x 128(E); 8 waves, each 32x64 (2x4 WMMA tiles).
// ---------------------------------------------------------------------------
__global__ __launch_bounds__(256) void qkv_gemm_kernel(
        const float* __restrict__ X, const float* __restrict__ W,
        unsigned short* __restrict__ Qw, unsigned short* __restrict__ Kw,
        unsigned short* __restrict__ Vw) {
    __shared__ unsigned Xs[128 * 16];   // 128 rows x 32 bf16 (16 dwords/row)
    __shared__ unsigned Ws[128 * 16];   // 128 rows x 32 bf16

    const int tid  = threadIdx.x;
    const int lane = tid & 31, w = tid >> 5;
    const int hi = lane >> 4, ln = lane & 15;
    const int m0 = blockIdx.y * 128;
    const int e0 = blockIdx.x * 128;
    const int mt = w & 3;               // 0..3: 32-row strip
    const int nt = w >> 2;              // 0..1: 64-col strip

    v8f acc[2][4] = {};

    for (int k0 = 0; k0 < 1024; k0 += 32) {
        __syncthreads();
        // X tile: 128x32 fp32 -> bf16 (2048 dwords, 8 per thread)
        #pragma unroll
        for (int i = 0; i < 8; ++i) {
            int lin = tid + 256 * i;
            int r = lin >> 4, c = lin & 15;
            const float* p = X + (size_t)(m0 + r) * 1024 + k0 + c * 2;
            Xs[lin] = pack2bf(p[0], p[1]);
        }
        // W tile: 128x32 fp32 -> bf16 (2048 dwords, 8 per thread)
        #pragma unroll
        for (int i = 0; i < 8; ++i) {
            int lin = tid + 256 * i;
            int r = lin >> 4, c = lin & 15;
            const float* p = W + (size_t)(e0 + r) * 1024 + k0 + c * 2;
            Ws[lin] = pack2bf(p[0], p[1]);
        }
        __syncthreads();

        FragAB a[2], bfr[4];
        #pragma unroll
        for (int i = 0; i < 2; ++i) {
            int row = mt * 32 + i * 16 + ln;       // A: lane = M row
            #pragma unroll
            for (int v = 0; v < 8; ++v) a[i].u[v] = Xs[row * 16 + a_kdw(v, hi)];
        }
        #pragma unroll
        for (int j = 0; j < 4; ++j) {
            int row = nt * 64 + j * 16 + ln;       // B: lane = N col = W row
            #pragma unroll
            for (int v = 0; v < 8; ++v) bfr[j].u[v] = Ws[row * 16 + b_kdw(v, hi)];
        }
        #pragma unroll
        for (int i = 0; i < 2; ++i)
            #pragma unroll
            for (int j = 0; j < 4; ++j)
                acc[i][j] = __builtin_amdgcn_wmma_f32_16x16x32_bf16(
                    false, a[i].v, false, bfr[j].v, (short)0, acc[i][j],
                    false, false);
    }

    // Scatter to Q/K/V [b,h,n,d] bf16. e0 is a multiple of 128 so the whole
    // workgroup maps to one of q/k/v; head varies with the upper column bit.
    const int kqv = e0 >> 10;
    const int h0  = (e0 >> 6) & 15;
    unsigned short* dst = (kqv == 0) ? Qw : ((kqv == 1) ? Kw : Vw);
    const float scl = (kqv == 0) ? 0.03125f : 1.0f;   // fold H^-0.5 into Q
    #pragma unroll
    for (int i = 0; i < 2; ++i) {
        #pragma unroll
        for (int j = 0; j < 4; ++j) {
            int dcol = nt * 64 + j * 16 + ln;         // C: lane = N col
            int h = h0 + (dcol >> 6);
            int d = dcol & 63;
            #pragma unroll
            for (int v = 0; v < 8; ++v) {             // C: row = v + 8*hi
                int m = m0 + mt * 32 + i * 16 + v + 8 * hi;
                int bb = m >> 11, n = m & 2047;
                dst[(((size_t)(bb * 16 + h)) * 2048 + n) * 64 + d] =
                    f2bf(acc[i][j][v] * scl);
            }
        }
    }
}

// ---------------------------------------------------------------------------
// Kernel 2: flash attention. Grid (N/128, NH, B); 8 waves x 16 rows each.
// K tile staged via Tensor Data Mover; V tile transposed via DS stores.
// ---------------------------------------------------------------------------
__global__ __launch_bounds__(256) void flash_attn_kernel(
        const unsigned short* __restrict__ Qw,
        const unsigned short* __restrict__ Kw,
        const unsigned short* __restrict__ Vw,
        const int* __restrict__ mask, float* __restrict__ out) {
    __shared__ unsigned KsU[64 * 32];       // K[j][d]  bf16 (8 KB), TDM dest
    __shared__ unsigned VtU[64 * 32];       // V^T[d][j] bf16 (8 KB)
    __shared__ unsigned PsU[8 * 16 * 32];   // per-wave P 16x64 bf16 (16 KB)
    unsigned short* VtH = (unsigned short*)VtU;
    unsigned short* PsH = (unsigned short*)PsU;

    const int tid  = threadIdx.x;
    const int lane = tid & 31, w = tid >> 5;
    const int hi = lane >> 4, ln = lane & 15;
    const int b = blockIdx.z, h = blockIdx.y;
    const int r0 = blockIdx.x * 128 + w * 16;

    const size_t headBase = ((size_t)(b * 16 + h)) * 2048 * 64;
    const unsigned*       Qu = (const unsigned*)(Qw + headBase);
    const unsigned short* Kh = Kw + headBase;
    const unsigned short* Vh = Vw + headBase;
    const unsigned ldsKs = (unsigned)(size_t)&KsU[0];

    // Q fragments for this wave's 16 rows (DH=64 -> two K=32 fragments)
    FragAB qf[2];
    {
        const unsigned* qrow = Qu + (size_t)(r0 + ln) * 32;
        #pragma unroll
        for (int ks = 0; ks < 2; ++ks)
            #pragma unroll
            for (int v = 0; v < 8; ++v)
                qf[ks].u[v] = qrow[a_kdw(v, hi) + 16 * ks];
    }

    // Query-row mask (reference masks whole rows with -10000)
    bool ok[8];
    #pragma unroll
    for (int v = 0; v < 8; ++v)
        ok[v] = mask[b * 2048 + r0 + v + 8 * hi] != 0;

    v8f accO[4] = {};
    float mrun[8], lrun[8];
    #pragma unroll
    for (int v = 0; v < 8; ++v) { mrun[v] = -3.0e38f; lrun[v] = 0.f; }

    for (int j0 = 0; j0 < 2048; j0 += 64) {
        __syncthreads();

        // K block 64x64 bf16: Tensor Data Mover global->LDS (wave 0 issues;
        // TDM ignores EXEC, descriptors are uniform -> SGPRs).
        if (w == 0) {
            unsigned long long ga =
                (unsigned long long)(const void*)(Kh + (size_t)j0 * 64);
            u32x4 g0 = { 1u,                                   // count=1
                         ldsKs,                                // lds_addr
                         (unsigned)ga,                         // global_addr lo
                         (unsigned)((ga >> 32) & 0x01FFFFFFu)  // addr[56:32]
                             | (2u << 30) };                   // type=2 (image)
            i32x8 g1 = { 0x00010000,          // data_size=2B
                         (int)(64u << 16),    // tensor_dim0 = 64
                         (int)(64u << 16),    // tensor_dim1 = 64
                         (int)(64u << 16),    // tile_dim0 = 64
                         64,                  // tile_dim1 = 64
                         64,                  // tensor_dim0_stride = 64
                         0, 0 };
            i32x4 gz = { 0, 0, 0, 0 };
#if __clang_major__ >= 23
            i32x8 gz8 = { 0, 0, 0, 0, 0, 0, 0, 0 };
            __builtin_amdgcn_tensor_load_to_lds(g0, g1, gz, gz, gz8, 0);
#else
            __builtin_amdgcn_tensor_load_to_lds(g0, g1, gz, gz, 0);
#endif
        }

        // V block transposed: Vt[d][j] (manual: TDM cannot transpose)
        #pragma unroll
        for (int i = 0; i < 16; ++i) {
            int lin = tid + 256 * i;
            int j = lin >> 6, d = lin & 63;
            VtH[d * 64 + j] = Vh[(size_t)(j0 + j) * 64 + d];
        }
        if (j0 + 64 < 2048)
            __builtin_prefetch(Vh + (size_t)(j0 + 64) * 64 + lane * 2, 0, 1);

        if (w == 0) __builtin_amdgcn_s_wait_tensorcnt(0);
        __syncthreads();

        // S = Q * K^T : 4 key subtiles x (2 x k32) WMMAs
        v8f s[4];
        #pragma unroll
        for (int t = 0; t < 4; ++t) {
            v8f st = {};
            #pragma unroll
            for (int ks = 0; ks < 2; ++ks) {
                FragAB kb;
                int j = t * 16 + ln;                   // B: lane = key col
                #pragma unroll
                for (int v = 0; v < 8; ++v)
                    kb.u[v] = KsU[j * 32 + b_kdw(v, hi) + 16 * ks];
                st = __builtin_amdgcn_wmma_f32_16x16x32_bf16(
                    false, qf[ks].v, false, kb.v, (short)0, st, false, false);
            }
            s[t] = st;
        }

        // Online softmax per row (row = v + 8*hi; 16-lane row groups)
        #pragma unroll
        for (int v = 0; v < 8; ++v) {
            float s0 = ok[v] ? s[0][v] : -10000.f;
            float s1 = ok[v] ? s[1][v] : -10000.f;
            float s2 = ok[v] ? s[2][v] : -10000.f;
            float s3 = ok[v] ? s[3][v] : -10000.f;
            float mloc = fmaxf(fmaxf(s0, s1), fmaxf(s2, s3));
            mloc = fmaxf(mloc, __shfl_xor(mloc, 1, 32));
            mloc = fmaxf(mloc, __shfl_xor(mloc, 2, 32));
            mloc = fmaxf(mloc, __shfl_xor(mloc, 4, 32));
            mloc = fmaxf(mloc, __shfl_xor(mloc, 8, 32));
            float mnew = fmaxf(mrun[v], mloc);
            float corr = __expf(mrun[v] - mnew);
            mrun[v] = mnew;
            float p0 = __expf(s0 - mnew), p1 = __expf(s1 - mnew);
            float p2 = __expf(s2 - mnew), p3 = __expf(s3 - mnew);
            float ps = p0 + p1 + p2 + p3;
            ps += __shfl_xor(ps, 1, 32);
            ps += __shfl_xor(ps, 2, 32);
            ps += __shfl_xor(ps, 4, 32);
            ps += __shfl_xor(ps, 8, 32);
            lrun[v] = lrun[v] * corr + ps;
            #pragma unroll
            for (int dt = 0; dt < 4; ++dt) accO[dt][v] *= corr;
            int row = v + 8 * hi;
            PsH[w * 1024 + row * 64 +  0 + ln] = f2bf(p0);
            PsH[w * 1024 + row * 64 + 16 + ln] = f2bf(p1);
            PsH[w * 1024 + row * 64 + 32 + ln] = f2bf(p2);
            PsH[w * 1024 + row * 64 + 48 + ln] = f2bf(p3);
        }

        // Same-wave cross-lane LDS store -> load: make DS ordering explicit.
        asm volatile("s_wait_dscnt 0" ::: "memory");

        // O += P * V : 4 d-subtiles x (2 x k32) WMMAs
        #pragma unroll
        for (int dt = 0; dt < 4; ++dt) {
            #pragma unroll
            for (int ks = 0; ks < 2; ++ks) {
                FragAB pa, vb;
                #pragma unroll
                for (int v = 0; v < 8; ++v) {
                    pa.u[v] = PsU[w * 512 + ln * 32 + a_kdw(v, hi) + 16 * ks];
                    vb.u[v] = VtU[(dt * 16 + ln) * 32 + b_kdw(v, hi) + 16 * ks];
                }
                accO[dt] = __builtin_amdgcn_wmma_f32_16x16x32_bf16(
                    false, pa.v, false, vb.v, (short)0, accO[dt], false, false);
            }
        }
    }

    // Normalize and store fp32 output [b, n, h*64 + d]
    float inv[8];
    #pragma unroll
    for (int v = 0; v < 8; ++v) inv[v] = 1.0f / lrun[v];
    #pragma unroll
    for (int dt = 0; dt < 4; ++dt) {
        int d = dt * 16 + ln;
        #pragma unroll
        for (int v = 0; v < 8; ++v) {
            int row = r0 + v + 8 * hi;
            out[((size_t)(b * 2048 + row)) * 1024 + h * 64 + d] =
                accO[dt][v] * inv[v];
        }
    }
}

// ---------------------------------------------------------------------------
extern "C" void kernel_launch(void* const* d_in, const int* in_sizes, int n_in,
                              void* d_out, int out_size, void* d_ws,
                              size_t ws_size, hipStream_t stream) {
    const float* X    = (const float*)d_in[0];   // hidden_states [2,2048,1024]
    const int*   mask = (const int*)d_in[1];     // attention_mask [2,2048]
    const float* W    = (const float*)d_in[2];   // W_qkv [3072,1024]
    float* out = (float*)d_out;                  // [2,2048,1024] fp32

    // Workspace: Q,K,V bf16 in [b,h,n,d]; 3 * 8 MB = 24 MB
    const size_t headElems = (size_t)2 * 16 * 2048 * 64;
    unsigned short* Qw = (unsigned short*)d_ws;
    unsigned short* Kw = Qw + headElems;
    unsigned short* Vw = Kw + headElems;

    qkv_gemm_kernel<<<dim3(24, 32), 256, 0, stream>>>(X, W, Qw, Kw, Vw);
    flash_attn_kernel<<<dim3(16, 16, 2), 256, 0, stream>>>(Qw, Kw, Vw, mask, out);
}
